// CircuitGenieModel_43353399886348
// MI455X (gfx1250) — matrix operand order repaired and checked by reading, not compiled, not run on hardware
//
#include <hip/hip_runtime.h>
#include <math.h>

// ---------------------------------------------------------------------------
// Types for CDNA5 WMMA (wave32): A/B = 16 bf16 per lane, C/D = 8 f32 per lane
// ---------------------------------------------------------------------------
typedef __attribute__((ext_vector_type(16))) __bf16        bf16x16;
typedef __attribute__((ext_vector_type(8)))  float         f32x8;
typedef __attribute__((ext_vector_type(4)))  unsigned int  u32x4;
typedef int gvec_i32x4 __attribute__((vector_size(16)));   // matches builtin sig

union FragAB { bf16x16 v; u32x4 q[2]; };

static __device__ inline __bf16 f2bf(float f) {
    union { float f; unsigned int u; } in; in.f = f;
    unsigned int r = in.u + 0x7FFFu + ((in.u >> 16) & 1u);   // RNE
    union { unsigned short u; __bf16 b; } out;
    out.u = (unsigned short)(r >> 16);
    return out.b;
}

static __device__ inline f32x8 wmma_bf16(const FragAB& a, const FragAB& b, f32x8 c) {
    return __builtin_amdgcn_wmma_f32_16x16x32_bf16(false, a.v, false, b.v,
                                                   (short)0, c, false, false);
}

// ---- CDNA5 async global->LDS copy (ASYNCcnt path), guarded fallback -------
#if __has_builtin(__builtin_amdgcn_global_load_async_to_lds_b128)
#define HAVE_ASYNC 1
#else
#define HAVE_ASYNC 0
#endif

static __device__ inline void g2l_b128(const __bf16* g, __bf16* l) {
#if HAVE_ASYNC
    __builtin_amdgcn_global_load_async_to_lds_b128(
        (__attribute__((address_space(1))) gvec_i32x4*)(void*)g,
        (__attribute__((address_space(3))) gvec_i32x4*)(void*)l, 0, 0);
#else
    *(u32x4*)l = *(const u32x4*)g;
#endif
}
static __device__ inline void g2l_wait() {
#if HAVE_ASYNC
#if __has_builtin(__builtin_amdgcn_s_wait_asynccnt)
    __builtin_amdgcn_s_wait_asynccnt(0);
#else
    asm volatile("s_wait_asynccnt 0" ::: "memory");
#endif
#endif
}

// ---------------------------------------------------------------------------
// Embedding: x[b,t,:] = token_emb[id] + pos_emb[t]   (f32 residual stream)
// ---------------------------------------------------------------------------
__global__ __launch_bounds__(256)
void embed_kernel(const int* __restrict__ ids, const float* __restrict__ tok,
                  const float* __restrict__ pos, float* __restrict__ x,
                  int T, int D) {
    int row = blockIdx.x;
    int t   = row % T;
    int id  = ids[row];
    int c0  = threadIdx.x * 4;
    const float* tp = tok + (size_t)id * D;
    const float* pp = pos + (size_t)t  * D;
    float*       xp = x   + (size_t)row * D;
#pragma unroll
    for (int j = 0; j < 4; ++j) xp[c0 + j] = tp[c0 + j] + pp[c0 + j];
}

// ---------------------------------------------------------------------------
// LayerNorm over D=1024 -> bf16 output (GEMM A-operand)
// ---------------------------------------------------------------------------
__global__ __launch_bounds__(256)
void ln_kernel(const float* __restrict__ x, const float* __restrict__ w,
               const float* __restrict__ b, __bf16* __restrict__ o, int D) {
    int row = blockIdx.x;
    int tid = threadIdx.x, lane = tid & 31, wid = tid >> 5;
    const float* xr = x + (size_t)row * D;
    float v[4], s = 0.f, ss = 0.f;
#pragma unroll
    for (int j = 0; j < 4; ++j) { v[j] = xr[tid * 4 + j]; s += v[j]; ss += v[j] * v[j]; }
#pragma unroll
    for (int off = 16; off >= 1; off >>= 1) {
        s  += __shfl_xor(s,  off, 32);
        ss += __shfl_xor(ss, off, 32);
    }
    __shared__ float red[16];
    __shared__ float bc[2];
    if (lane == 0) { red[wid] = s; red[8 + wid] = ss; }
    __syncthreads();
    if (tid == 0) {
        float S = 0.f, SS = 0.f;
        for (int i = 0; i < 8; ++i) { S += red[i]; SS += red[8 + i]; }
        float mu  = S / (float)D;
        float var = SS / (float)D - mu * mu;
        bc[0] = mu;
        bc[1] = rsqrtf(var + 1e-5f);
    }
    __syncthreads();
    float mu = bc[0], rstd = bc[1];
    __bf16* orow = o + (size_t)row * D;
#pragma unroll
    for (int j = 0; j < 4; ++j) {
        int c = tid * 4 + j;
        orow[c] = f2bf((v[j] - mu) * rstd * w[c] + b[c]);
    }
}

// ---------------------------------------------------------------------------
// Weight transpose+convert: W[K][N] f32 -> WT[N][K] bf16 (32x32 tiles)
// ---------------------------------------------------------------------------
__global__ __launch_bounds__(256)
void tcvt_kernel(const float* __restrict__ W, __bf16* __restrict__ WT,
                 int K, int N) {
    __shared__ float t[32][33];
    int n0 = blockIdx.x * 32, k0 = blockIdx.y * 32;
    int tx = threadIdx.x & 31, ty = threadIdx.x >> 5;   // 8 rows per pass
#pragma unroll
    for (int i = 0; i < 4; ++i) {
        int k = ty + i * 8;
        t[k][tx] = W[(size_t)(k0 + k) * N + n0 + tx];
    }
    __syncthreads();
#pragma unroll
    for (int i = 0; i < 4; ++i) {
        int n = ty + i * 8;
        WT[(size_t)(n0 + n) * K + k0 + tx] = f2bf(t[tx][n]);
    }
}

// Plain convert (token_emb is already [N][K] for the tied head)
__global__ __launch_bounds__(256)
void cvt_kernel(const float* __restrict__ s, __bf16* __restrict__ d, long n) {
    for (long i = (long)blockIdx.x * 256 + threadIdx.x; i < n;
         i += (long)gridDim.x * 256)
        d[i] = f2bf(s[i]);
}

// ---------------------------------------------------------------------------
// GEMM: out = act(A * Bt^T + bias) (+res).  A: bf16 [M,K], Bt: bf16 [N,K].
// Block tile 128x128x32, 8 waves, wave tile 32x64 -> 8 WMMA per K-step.
// Staging = pure 16B copies (async-to-LDS when available).
// outB != null -> bf16 out; else f32 out (+ optional residual).
// ---------------------------------------------------------------------------
#define GLDS 40   // padded LDS row stride (elements); 80 B, multiple of 16 B
__global__ __launch_bounds__(256)
void gemm_kernel(const __bf16* __restrict__ A, const __bf16* __restrict__ Bt,
                 const float* __restrict__ bias, const float* res,
                 float* outF, __bf16* outB,
                 int M, int N, int K, int actGelu) {
    __shared__ __bf16 sA[128 * GLDS];
    __shared__ __bf16 sB[128 * GLDS];   // [n][k]
    const int tid  = threadIdx.x;
    const int lane = tid & 31, wid = tid >> 5;
    const int lr   = lane & 15, lh = lane >> 4;
    const int wm   = (wid & 3) * 32;
    const int wn   = (wid >> 2) * 64;
    const int m0   = blockIdx.y * 128;
    const int n0   = blockIdx.x * 128;

    f32x8 acc[2][4];
#pragma unroll
    for (int mt = 0; mt < 2; ++mt)
#pragma unroll
        for (int nt = 0; nt < 4; ++nt)
#pragma unroll
            for (int i = 0; i < 8; ++i) acc[mt][nt][i] = 0.f;

    for (int k0 = 0; k0 < K; k0 += 32) {
        __syncthreads();
        // ---- stage A and B tiles: 512 x 16B copies each ----
#pragma unroll
        for (int i = 0; i < 2; ++i) {
            int idx = tid + i * 256;
            int r = idx >> 2, c = (idx & 3) << 3;
            g2l_b128(A  + (size_t)(m0 + r) * K + k0 + c, &sA[r * GLDS + c]);
            g2l_b128(Bt + (size_t)(n0 + r) * K + k0 + c, &sB[r * GLDS + c]);
        }
        g2l_wait();
        __syncthreads();

        FragAB fa[2], fb[4];
#pragma unroll
        for (int mt = 0; mt < 2; ++mt) {
            // A 16x32: lane m=lr; elems 0..7 -> k=8*lh+e ; 8..15 -> k=16+8*lh+e
            const __bf16* p = &sA[(wm + mt * 16 + lr) * GLDS + lh * 8];
            fa[mt].q[0] = *(const u32x4*)(p);
            fa[mt].q[1] = *(const u32x4*)(p + 16);
        }
#pragma unroll
        for (int nt = 0; nt < 4; ++nt) {
            // B 32x16: lane n=lr; elems 0..7 -> k=8*lh+e ; 8..15 -> k=16+8*lh+e
            const __bf16* p = &sB[(wn + nt * 16 + lr) * GLDS + lh * 8];
            fb[nt].q[0] = *(const u32x4*)(p);
            fb[nt].q[1] = *(const u32x4*)(p + 16);
        }
#pragma unroll
        for (int mt = 0; mt < 2; ++mt)
#pragma unroll
            for (int nt = 0; nt < 4; ++nt)
                acc[mt][nt] = wmma_bf16(fa[mt], fb[nt], acc[mt][nt]);
    }

    // ---- epilogue: C rows m = r + 8*lh, cols n = lr ----
#pragma unroll
    for (int mt = 0; mt < 2; ++mt) {
        int gmb = m0 + wm + mt * 16 + lh * 8;
#pragma unroll
        for (int nt = 0; nt < 4; ++nt) {
            int gn = n0 + wn + nt * 16 + lr;
            float bv = bias ? bias[gn] : 0.f;
#pragma unroll
            for (int r = 0; r < 8; ++r) {
                size_t g = (size_t)(gmb + r) * N + gn;
                float v = acc[mt][nt][r] + bv;
                if (actGelu) v = 0.5f * v * (1.f + erff(v * 0.70710678118f));
                if (outB) {
                    outB[g] = f2bf(v);
                } else {
                    if (res) v += res[g];
                    outF[g] = v;
                }
            }
        }
    }
}

// ---------------------------------------------------------------------------
// Flash attention (causal, HD=64), qkv bf16 [M][3072] -> att bf16 [M][1024].
// Block = 64 q-rows of one (b,h); 4 waves x 16 rows.
// ---------------------------------------------------------------------------
__global__ __launch_bounds__(128)
void attn_kernel(const __bf16* __restrict__ qkv, __bf16* __restrict__ out,
                 int B, int T, int H) {
    const int D3 = 3072, Dm = 1024;
    __shared__ __bf16 sK[32 * 72];          // [key][hd]
    __shared__ __bf16 sV[64 * 40];          // [hd][key] (transposed)
    __shared__ __bf16 sP[4 * 16 * 40];      // per-wave P scratch [m][k]

    const int tiles = T / 64;
    const int qt = blockIdx.x % tiles;
    const int bh = blockIdx.x / tiles;
    const int h  = bh % H, b = bh / H;
    const int tid = threadIdx.x, lane = tid & 31, w = tid >> 5;
    const int lr = lane & 15, lh = lane >> 4;
    const int q0 = qt * 64;

    // ---- Q fragments: direct 16B loads from bf16 HBM ----
    FragAB fq[2];
    {
        const __bf16* qp = qkv + (size_t)(b * T + q0 + w * 16 + lr) * D3 + h * 64;
#pragma unroll
        for (int c = 0; c < 2; ++c) {
            fq[c].q[0] = *(const u32x4*)(qp + c * 32 + lh * 8);
            fq[c].q[1] = *(const u32x4*)(qp + c * 32 + 16 + lh * 8);
        }
    }

    float mrow[8], lrow[8];
    f32x8 o[4];
#pragma unroll
    for (int r = 0; r < 8; ++r) { mrow[r] = -1e30f; lrow[r] = 0.f; }
#pragma unroll
    for (int nt = 0; nt < 4; ++nt)
#pragma unroll
        for (int r = 0; r < 8; ++r) o[nt][r] = 0.f;

    const int nsteps = (q0 + 64) / 32;
    for (int s = 0; s < nsteps; ++s) {
        const int kb = s * 32;
        __syncthreads();
        // ---- stage K (direct rows) and V (transposed) ----
#pragma unroll
        for (int i = 0; i < 2; ++i) {
            int idx = tid + i * 128;
            int key = idx >> 3, c = (idx & 7) << 3;
            const __bf16* base = qkv + (size_t)(b * T + kb + key) * D3 + h * 64;
            *(u32x4*)&sK[key * 72 + c] = *(const u32x4*)(base + 1024 + c);
            u32x4 raw = *(const u32x4*)(base + 2048 + c);
            const __bf16* e = (const __bf16*)&raw;
#pragma unroll
            for (int j = 0; j < 8; ++j) sV[(c + j) * 40 + key] = e[j];
        }
        __syncthreads();

        // ---- S = (Q K^T) * scale ----
        f32x8 sf[2];
#pragma unroll
        for (int kt = 0; kt < 2; ++kt) {
            f32x8 sacc;
#pragma unroll
            for (int i = 0; i < 8; ++i) sacc[i] = 0.f;
#pragma unroll
            for (int c = 0; c < 2; ++c) {
                FragAB fk;
                const __bf16* p = &sK[(kt * 16 + lr) * 72 + c * 32 + lh * 8];
                fk.q[0] = *(const u32x4*)(p);
                fk.q[1] = *(const u32x4*)(p + 16);
                sacc = wmma_bf16(fq[c], fk, sacc);
            }
            sf[kt] = sacc;
        }
#pragma unroll
        for (int kt = 0; kt < 2; ++kt)
#pragma unroll
            for (int r = 0; r < 8; ++r) {
                float sv = sf[kt][r] * 0.125f;
                int key = kb + kt * 16 + lr;
                int q   = q0 + w * 16 + r + 8 * lh;
                sf[kt][r] = (key > q) ? -1e30f : sv;
            }
        // ---- online softmax (reduce over 16-lane half-groups) ----
        float mr[8];
#pragma unroll
        for (int r = 0; r < 8; ++r) mr[r] = fmaxf(sf[0][r], sf[1][r]);
#pragma unroll
        for (int off = 1; off <= 8; off <<= 1)
#pragma unroll
            for (int r = 0; r < 8; ++r) mr[r] = fmaxf(mr[r], __shfl_xor(mr[r], off, 32));
        float alpha[8], rs[8];
#pragma unroll
        for (int r = 0; r < 8; ++r) {
            float mn = fmaxf(mrow[r], mr[r]);
            alpha[r] = __expf(mrow[r] - mn);
            mrow[r]  = mn;
            float a0 = (sf[0][r] <= -1e29f) ? 0.f : __expf(sf[0][r] - mn);
            float a1 = (sf[1][r] <= -1e29f) ? 0.f : __expf(sf[1][r] - mn);
            int m = r + 8 * lh;
            sP[w * 640 + m * 40 + lr]      = f2bf(a0);
            sP[w * 640 + m * 40 + 16 + lr] = f2bf(a1);
            rs[r] = a0 + a1;
        }
#pragma unroll
        for (int off = 1; off <= 8; off <<= 1)
#pragma unroll
            for (int r = 0; r < 8; ++r) rs[r] += __shfl_xor(rs[r], off, 32);
#pragma unroll
        for (int r = 0; r < 8; ++r) lrow[r] = lrow[r] * alpha[r] + rs[r];
#pragma unroll
        for (int nt = 0; nt < 4; ++nt)
#pragma unroll
            for (int r = 0; r < 8; ++r) o[nt][r] *= alpha[r];

        asm volatile("s_wait_dscnt 0" ::: "memory");   // same-wave LDS RAW

        // ---- O += P * V ----
        FragAB fp;
        {
            const __bf16* p = &sP[w * 640 + lr * 40 + lh * 8];
            fp.q[0] = *(const u32x4*)(p);
            fp.q[1] = *(const u32x4*)(p + 16);
        }
#pragma unroll
        for (int nt = 0; nt < 4; ++nt) {
            FragAB fv;
            const __bf16* p = &sV[(nt * 16 + lr) * 40 + lh * 8];
            fv.q[0] = *(const u32x4*)(p);
            fv.q[1] = *(const u32x4*)(p + 16);
            o[nt] = wmma_bf16(fp, fv, o[nt]);
        }
    }

    // ---- normalize, store bf16 ----
#pragma unroll
    for (int r = 0; r < 8; ++r) {
        float inv = (lrow[r] > 0.f) ? 1.f / lrow[r] : 0.f;
        int q = q0 + w * 16 + r + 8 * lh;
        __bf16* op = out + (size_t)(b * T + q) * Dm + h * 64;
#pragma unroll
        for (int nt = 0; nt < 4; ++nt) op[nt * 16 + lr] = f2bf(o[nt][r] * inv);
    }
}

// ---------------------------------------------------------------------------
// Weighted CE loss over logits rows (V=4096)
// ---------------------------------------------------------------------------
__global__ __launch_bounds__(256)
void ce_kernel(const float* __restrict__ logits, const int* __restrict__ tgt,
               const unsigned char* __restrict__ vmask, float* accum, int V) {
    int row = blockIdx.x, tid = threadIdx.x, lane = tid & 31, wid = tid >> 5;
    const float* lg = logits + (size_t)row * V;
    __shared__ float sh[8];
    __shared__ float bval;
    float mx = -3.4e38f;
    for (int i = tid; i < V; i += 256) mx = fmaxf(mx, lg[i]);
#pragma unroll
    for (int off = 16; off >= 1; off >>= 1) mx = fmaxf(mx, __shfl_xor(mx, off, 32));
    if (lane == 0) sh[wid] = mx;
    __syncthreads();
    if (tid == 0) {
        float m = sh[0];
        for (int i = 1; i < 8; ++i) m = fmaxf(m, sh[i]);
        bval = m;
    }
    __syncthreads();
    mx = bval;
    float se = 0.f;
    for (int i = tid; i < V; i += 256) se += __expf(lg[i] - mx);
#pragma unroll
    for (int off = 16; off >= 1; off >>= 1) se += __shfl_xor(se, off, 32);
    __syncthreads();
    if (lane == 0) sh[wid] = se;
    __syncthreads();
    if (tid == 0) {
        float S = 0.f;
        for (int i = 0; i < 8; ++i) S += sh[i];
        int t = tgt[row];
        float nll = logf(S) + mx - lg[t];
        float wg  = 1.f + 4.f * (vmask[row] ? 1.f : 0.f);
        atomicAdd(accum + 0, (t == 0) ? 0.f : nll * wg);
        atomicAdd(accum + 1, wg);
    }
}

__global__ void zero_kernel(float* a) { if (threadIdx.x < 2) a[threadIdx.x] = 0.f; }
__global__ void fin_kernel(const float* a, float* out, unsigned long long off) {
    out[off] = a[0] / a[1];
}

// ---------------------------------------------------------------------------
extern "C" void kernel_launch(void* const* d_in, const int* in_sizes, int n_in,
                              void* d_out, int out_size, void* d_ws, size_t ws_size,
                              hipStream_t stream) {
    (void)n_in; (void)out_size; (void)ws_size;
    const int T = 2048, D = 1024, H = 16, DF = 4096, V = 4096, L = 8;
    const int M = in_sizes[0];                 // B*T = 8192

    const int*   ids   = (const int*)d_in[0];
    const int*   tgts  = (const int*)d_in[1];
    const unsigned char* vmask = (const unsigned char*)d_in[2];
    const float* tok   = (const float*)d_in[3];
    const float* pos   = (const float*)d_in[4];
    const float* ln1w  = (const float*)d_in[5];
    const float* ln1b  = (const float*)d_in[6];
    const float* wqkv  = (const float*)d_in[7];
    const float* bqkv  = (const float*)d_in[8];
    const float* wproj = (const float*)d_in[9];
    const float* bproj = (const float*)d_in[10];
    const float* ln2w  = (const float*)d_in[11];
    const float* ln2b  = (const float*)d_in[12];
    const float* w1    = (const float*)d_in[13];
    const float* b1    = (const float*)d_in[14];
    const float* w2    = (const float*)d_in[15];
    const float* b2    = (const float*)d_in[16];
    const float* lnfw  = (const float*)d_in[17];
    const float* lnfb  = (const float*)d_in[18];

    // ---- workspace layout (bytes) ----
    char* wsb = (char*)d_ws;
    size_t szX   = (size_t)M * D * 4;          // f32 residual stream
    size_t szQKV = (size_t)M * 3 * D * 2;      // bf16
    size_t szATT = (size_t)M * D * 2;          // bf16
    size_t szH   = (size_t)M * D * 2;          // bf16
    size_t szW   = (size_t)DF * D * 2;         // bf16 (largest per-layer weight)
    size_t szTok = (size_t)V * D * 2;          // bf16

    float*  x    = (float*)(wsb);
    __bf16* qkv  = (__bf16*)(wsb + szX);
    __bf16* att  = (__bf16*)(wsb + szX + szQKV);
    __bf16* ffh  = (__bf16*)(wsb + szX);       // aliases dead qkv+att (M*4D == M*DF)
    __bf16* hbuf = (__bf16*)(wsb + szX + szQKV + szATT);
    __bf16* wT   = (__bf16*)(wsb + szX + szQKV + szATT + szH);
    __bf16* tokb = (__bf16*)(wsb + szX + szQKV + szATT + szH + szW);
    float*  accum= (float*)(wsb + szX + szQKV + szATT + szH + szW + szTok);
    float*  logits = (float*)d_out;

    cvt_kernel<<<4096, 256, 0, stream>>>(tok, tokb, (long)V * D);
    embed_kernel<<<M, 256, 0, stream>>>(ids, tok, pos, x, T, D);

    for (int l = 0; l < L; ++l) {
        ln_kernel<<<M, 256, 0, stream>>>(x, ln1w + (size_t)l * D, ln1b + (size_t)l * D, hbuf, D);
        tcvt_kernel<<<dim3(3 * D / 32, D / 32), 256, 0, stream>>>(
            wqkv + (size_t)l * D * 3 * D, wT, D, 3 * D);
        gemm_kernel<<<dim3(3 * D / 128, M / 128), 256, 0, stream>>>(
            hbuf, wT, bqkv + (size_t)l * 3 * D, nullptr, nullptr, qkv,
            M, 3 * D, D, 0);
        attn_kernel<<<(M / T) * H * (T / 64), 128, 0, stream>>>(qkv, att, M / T, T, H);
        tcvt_kernel<<<dim3(D / 32, D / 32), 256, 0, stream>>>(
            wproj + (size_t)l * D * D, wT, D, D);
        gemm_kernel<<<dim3(D / 128, M / 128), 256, 0, stream>>>(
            att, wT, bproj + (size_t)l * D, x, x, nullptr, M, D, D, 0);
        ln_kernel<<<M, 256, 0, stream>>>(x, ln2w + (size_t)l * D, ln2b + (size_t)l * D, hbuf, D);
        tcvt_kernel<<<dim3(DF / 32, D / 32), 256, 0, stream>>>(
            w1 + (size_t)l * D * DF, wT, D, DF);
        gemm_kernel<<<dim3(DF / 128, M / 128), 256, 0, stream>>>(
            hbuf, wT, b1 + (size_t)l * DF, nullptr, nullptr, ffh, M, DF, D, 1);
        tcvt_kernel<<<dim3(D / 32, DF / 32), 256, 0, stream>>>(
            w2 + (size_t)l * DF * D, wT, DF, D);
        gemm_kernel<<<dim3(D / 128, M / 128), 256, 0, stream>>>(
            ffh, wT, b2 + (size_t)l * D, x, x, nullptr, M, D, DF, 0);
    }

    ln_kernel<<<M, 256, 0, stream>>>(x, lnfw, lnfb, hbuf, D);
    gemm_kernel<<<dim3(V / 128, M / 128), 256, 0, stream>>>(
        hbuf, tokb, nullptr, nullptr, logits, nullptr, M, V, D, 0);

    zero_kernel<<<1, 32, 0, stream>>>(accum);
    ce_kernel<<<M, 256, 0, stream>>>(logits, tgts, vmask, accum, V);
    fin_kernel<<<1, 1, 0, stream>>>(accum, logits, (unsigned long long)M * V);
}
